// MHA_23897198035374
// MI455X (gfx1250) — matrix operand order repaired and checked by reading, not compiled, not run on hardware
//
#include <hip/hip_runtime.h>
#include <hip/hip_bf16.h>

// MI455X (gfx1250) fused MHA:
//   k1: QKV projection via v_wmma_f32_16x16x32_f16, software-pipelined
//       global->reg->LDS transposed staging, q/k row-norm (ddof=1, eps on std),
//       scale 1/8 * log2(e) folded into Q, V stored transposed (e-major).
//   k2: flash attention, 16 query rows per wave, 64-key chunks, base-2 online
//       softmax (v_exp_f32 via builtin), K/V tiles double-buffered in LDS via
//       global_load_async_to_lds_b128 (ASYNCcnt) with one barrier per chunk,
//       P C->A layout via per-wave LDS roundtrip, O written transposed.
// Workspace: 3 * 16 MB f16 (Q, K, Vt) = 48 MB.

typedef __attribute__((ext_vector_type(16))) _Float16 v16h;
typedef __attribute__((ext_vector_type(8)))  _Float16 v8h;
typedef __attribute__((ext_vector_type(8)))  float    v8f;
typedef __attribute__((ext_vector_type(4)))  float    v4f;

#define B_SZ 4
#define H_SZ 16
#define T_SZ 2048
#define D_SZ 1024
#define DH   64

static __device__ __forceinline__ v8f wmma_f16(v16h a, v16h b, v8f c) {
  // D = A(16x32 f16) * B(32x16 f16) + C(16x16 f32)
  return __builtin_amdgcn_wmma_f32_16x16x32_f16(false, a, false, b,
                                                (short)0, c, false, false);
}

static __device__ __forceinline__ v16h pack16(v8h lo, v8h hi) {
  return __builtin_shufflevector(lo, hi, 0, 1, 2, 3, 4, 5, 6, 7,
                                 8, 9, 10, 11, 12, 13, 14, 15);
}

// CDNA5 async copy: global -> LDS, 16 bytes per lane, tracked by ASYNCcnt.
static __device__ __forceinline__ void async_copy_b128(unsigned lds_off,
                                                       const void* gptr) {
  asm volatile("global_load_async_to_lds_b128 %0, %1, off"
               :: "v"(lds_off),
                  "v"((unsigned long long)(uintptr_t)gptr)
               : "memory");
}
static __device__ __forceinline__ void wait_async0() {
  asm volatile("s_wait_asynccnt 0x0" ::: "memory");
}

// ---------------------------------------------------------------------------
// Kernel 1: h[n,b,h,t,e] = sum_d x[b,d,t] * qkv[n,h,d,e]; norm q,k; stash.
// grid = 3*B*H*(T/128) blocks of 256 threads; each wave owns 16 t-rows x 64 e.
// ---------------------------------------------------------------------------
__global__ __launch_bounds__(256) void qkv_proj_norm_kernel(
    const float* __restrict__ x, const float* __restrict__ qkv,
    _Float16* __restrict__ qws, _Float16* __restrict__ kws,
    _Float16* __restrict__ vtws) {
  __shared__ __align__(16) _Float16 XT[128][40]; // [t-local][d-local], pad 40
  __shared__ __align__(16) _Float16 WT[64][40];  // [e][d-local], pad 40

  const int tid  = threadIdx.x;
  const int lane = tid & 31;
  const int wave = tid >> 5;
  const int half = lane >> 4;   // 0: rows/K-low group, 1: high group
  const int r16  = lane & 15;

  const int blk   = blockIdx.x;
  const int tblk  = blk & 15;         // T/128 = 16 tiles
  const int h     = (blk >> 4) & 15;
  const int b     = (blk >> 8) & 3;
  const int n     = blk >> 10;        // 0=q 1=k 2=v
  const int tBase = tblk * 128;

  v8f acc[4] = {{}, {}, {}, {}};

  const int dr = tid >> 3;            // 0..31 (d within chunk)
  const int tq = (tid & 7) * 16;      // t sub-offset for X stage
  const int e8 = (tid & 7) * 8;       // e sub-offset for W stage

  const float* xbase = x + (size_t)b * D_SZ * T_SZ + tBase + tq;
  const float* wbase = qkv + (((size_t)n * H_SZ + h) * D_SZ) * DH + e8;

  // software pipeline: preload chunk 0 into registers
  v4f xv[4], wv[2];
  {
    const v4f* xp = (const v4f*)(xbase + (size_t)dr * T_SZ);
    xv[0] = xp[0]; xv[1] = xp[1]; xv[2] = xp[2]; xv[3] = xp[3];
    const v4f* wp = (const v4f*)(wbase + (size_t)dr * DH);
    wv[0] = wp[0]; wv[1] = wp[1];
  }

  for (int d0 = 0; d0 < D_SZ; d0 += 32) {
    // --- commit staged registers to LDS (transposed, f16) ---
#pragma unroll
    for (int i = 0; i < 4; ++i)
#pragma unroll
      for (int j = 0; j < 4; ++j)
        XT[tq + i * 4 + j][dr] = (_Float16)xv[i][j];
#pragma unroll
    for (int j = 0; j < 4; ++j) {
      WT[e8 + j][dr]     = (_Float16)wv[0][j];
      WT[e8 + 4 + j][dr] = (_Float16)wv[1][j];
    }
    __syncthreads();

    // --- preload next chunk; load latency overlaps the WMMAs below ---
    if (d0 + 32 < D_SZ) {
      const v4f* xp = (const v4f*)(xbase + (size_t)(d0 + 32 + dr) * T_SZ);
      xv[0] = xp[0]; xv[1] = xp[1]; xv[2] = xp[2]; xv[3] = xp[3];
      const v4f* wp = (const v4f*)(wbase + (size_t)(d0 + 32 + dr) * DH);
      wv[0] = wp[0]; wv[1] = wp[1];
    }

    // A fragment: lane<16 -> M=lane, K {0..7,16..23}; lane>=16 -> K {8..15,24..31}
    const _Float16* arow = &XT[wave * 16 + r16][0];
    v16h a = pack16(*(const v8h*)(arow + half * 8),
                    *(const v8h*)(arow + 16 + half * 8));
#pragma unroll
    for (int f = 0; f < 4; ++f) {
      // B fragment: lane<16 -> N=lane, K {0..7,16..23}; lane>=16 -> K {8..15,24..31}
      const _Float16* brow = &WT[f * 16 + r16][half * 8];
      v16h bb = pack16(*(const v8h*)(brow), *(const v8h*)(brow + 16));
      acc[f] = wmma_f16(a, bb, acc[f]);
    }
    __syncthreads();
  }

  const size_t bh  = (size_t)b * H_SZ + h;
  const int    t0w = tBase + wave * 16;

  if (n < 2) {
    // row-wise mean / unbiased std over e (64) ; C row r lives in v8f slot r
    float mean[8], inv[8];
#pragma unroll
    for (int r = 0; r < 8; ++r) {
      float s = acc[0][r] + acc[1][r] + acc[2][r] + acc[3][r];
      float q = acc[0][r] * acc[0][r] + acc[1][r] * acc[1][r] +
                acc[2][r] * acc[2][r] + acc[3][r] * acc[3][r];
#pragma unroll
      for (int mask = 1; mask < 16; mask <<= 1) {
        s += __shfl_xor(s, mask, 32);
        q += __shfl_xor(q, mask, 32);
      }
      float m   = s * (1.0f / 64.0f);
      float var = (q - 64.0f * m * m) * (1.0f / 63.0f);
      var = var < 0.0f ? 0.0f : var;
      float sd = sqrtf(var);
      mean[r] = m;
      // Q gets 1/sqrt(64) * log2(e) so kernel 2 can run base-2 softmax.
      inv[r] = ((n == 0) ? 0.125f * 1.44269504088896f : 1.0f) / (sd + 1e-5f);
    }
    _Float16* dst = (n == 0) ? qws : kws;
#pragma unroll
    for (int f = 0; f < 4; ++f)
#pragma unroll
      for (int r = 0; r < 8; ++r) {
        float v = (acc[f][r] - mean[r]) * inv[r];
        dst[(bh * T_SZ + (size_t)(t0w + half * 8 + r)) * DH + f * 16 + r16] =
            (_Float16)v;
      }
  } else {
    // V transposed store: lane owns (e = f*16+r16, t = t0w+half*8 .. +8) contiguously
#pragma unroll
    for (int f = 0; f < 4; ++f) {
      v8h pk;
#pragma unroll
      for (int r = 0; r < 8; ++r) pk[r] = (_Float16)acc[f][r];
      *(v8h*)(vtws + (bh * DH + (size_t)(f * 16 + r16)) * T_SZ + t0w + half * 8) = pk;
    }
  }
}

// ---------------------------------------------------------------------------
// Kernel 2: flash attention. grid = B*H*(T/128), 256 threads; each wave owns a
// 16-query tile; 64-key chunks double-buffered in LDS via async copies.
// ---------------------------------------------------------------------------
__global__ __launch_bounds__(256) void flash_attn_kernel(
    const _Float16* __restrict__ qws, const _Float16* __restrict__ kws,
    const _Float16* __restrict__ vtws, float* __restrict__ out) {
  __shared__ __align__(16) _Float16 Ktile[2][64][72]; // [buf][key][dim], pad 72
  __shared__ __align__(16) _Float16 Vtile[2][64][72]; // [buf][e][key-local]
  __shared__ __align__(16) _Float16 Pb[8][16][72];    // per-wave P roundtrip

  const int tid  = threadIdx.x;
  const int lane = tid & 31;
  const int wave = tid >> 5;
  const int half = lane >> 4;
  const int r16  = lane & 15;

  const int qblk = blockIdx.x & 15;
  const int h    = (blockIdx.x >> 4) & 15;
  const int b    = blockIdx.x >> 8;
  const size_t bh = (size_t)b * H_SZ + h;
  const int q0 = qblk * 128 + wave * 16;

  // Q A-fragments for dims 0..31 and 32..63 (held for the whole kernel)
  const _Float16* qrow = qws + (bh * T_SZ + (size_t)(q0 + r16)) * DH;
  v16h aq[2];
#pragma unroll
  for (int d = 0; d < 2; ++d)
    aq[d] = pack16(*(const v8h*)(qrow + d * 32 + half * 8),
                   *(const v8h*)(qrow + d * 32 + 16 + half * 8));

  v8f o[4] = {{}, {}, {}, {}};
  float mrow[8], lrow[8];
#pragma unroll
  for (int r = 0; r < 8; ++r) { mrow[r] = -1e30f; lrow[r] = 0.0f; }

  // Staging geometry: 64 rows x 64 f16 = 512 x 16B chunks per tile; 2 per thread.
  const unsigned kbase = (unsigned)(uintptr_t)&Ktile[0][0][0];
  const unsigned vbase = (unsigned)(uintptr_t)&Vtile[0][0][0];
  const unsigned tileBytes = 64u * 72u * 2u;     // 9216 B per buffer
  const _Float16* kgbl = kws + bh * T_SZ * DH;   // + (k0+row)*64 + col8
  const _Float16* vgbl = vtws + bh * DH * T_SZ;  // + row*T + k0 + col8

  auto stage = [&](int buf, int k0) {
#pragma unroll
    for (int i = 0; i < 2; ++i) {
      const int c    = tid + i * 256;      // 0..511 chunk id
      const int row  = c >> 3;             // 0..63
      const int col8 = (c & 7) * 8;        // f16 column offset
      const unsigned lofs = (unsigned)buf * tileBytes +
                            (unsigned)(row * 72 + col8) * 2u;
      async_copy_b128(kbase + lofs, kgbl + (size_t)(k0 + row) * DH + col8);
      async_copy_b128(vbase + lofs, vgbl + (size_t)row * T_SZ + k0 + col8);
    }
  };

  stage(0, 0); // prologue fill

  for (int it = 0; it < T_SZ / 64; ++it) {
    const int cur = it & 1;
    const int k0  = it * 64;

    wait_async0();       // this wave's copies for chunk `it` have landed
    __syncthreads();     // all waves' copies visible; prev chunk fully consumed

    if (it + 1 < T_SZ / 64) stage(1 - cur, k0 + 64); // overlap with compute

    // ---- scores S(16x64) = Q @ K^T : four 16-key subtiles, K=64 dims ----
    v8f cs[4];
#pragma unroll
    for (int s = 0; s < 4; ++s) {
      v8f c = {};
#pragma unroll
      for (int d = 0; d < 2; ++d) {
        const _Float16* kr = &Ktile[cur][s * 16 + r16][d * 32 + half * 8];
        v16h bk = pack16(*(const v8h*)(kr), *(const v8h*)(kr + 16));
        c = wmma_f16(aq[d], bk, c);
      }
      cs[s] = c;
    }

    // ---- base-2 online softmax over this chunk ----
    float p[4][8], corr[8];
#pragma unroll
    for (int r = 0; r < 8; ++r) {
      float mx = fmaxf(fmaxf(cs[0][r], cs[1][r]), fmaxf(cs[2][r], cs[3][r]));
#pragma unroll
      for (int mask = 1; mask < 16; mask <<= 1)
        mx = fmaxf(mx, __shfl_xor(mx, mask, 32));
      float mn = fmaxf(mrow[r], mx);
      corr[r] = __builtin_amdgcn_exp2f(mrow[r] - mn);
      mrow[r] = mn;
      float rs = 0.0f;
#pragma unroll
      for (int s = 0; s < 4; ++s) {
        p[s][r] = __builtin_amdgcn_exp2f(cs[s][r] - mn);
        rs += p[s][r];
      }
#pragma unroll
      for (int mask = 1; mask < 16; mask <<= 1)
        rs += __shfl_xor(rs, mask, 32);
      lrow[r] = lrow[r] * corr[r] + rs;
    }
#pragma unroll
    for (int f = 0; f < 4; ++f)
#pragma unroll
      for (int r = 0; r < 8; ++r) o[f][r] *= corr[r];

    // ---- P (C-layout) -> LDS row-major -> A-fragments (same-wave DS order) ----
#pragma unroll
    for (int s = 0; s < 4; ++s)
#pragma unroll
      for (int r = 0; r < 8; ++r)
        Pb[wave][half * 8 + r][s * 16 + r16] = (_Float16)p[s][r];
    const _Float16* prow = &Pb[wave][r16][0];
    v16h ap[2];
#pragma unroll
    for (int kk = 0; kk < 2; ++kk)
      ap[kk] = pack16(*(const v8h*)(prow + kk * 32 + half * 8),
                      *(const v8h*)(prow + kk * 32 + 16 + half * 8));

    // ---- O += P @ V : Vt rows give contiguous B-fragments ----
#pragma unroll
    for (int f = 0; f < 4; ++f)
#pragma unroll
      for (int kk = 0; kk < 2; ++kk) {
        const _Float16* vr = &Vtile[cur][f * 16 + r16][kk * 32 + half * 8];
        v16h bv = pack16(*(const v8h*)(vr), *(const v8h*)(vr + 16));
        o[f] = wmma_f16(ap[kk], bv, o[f]);
      }
  }

  // ---- epilogue: divide by l, store transposed out[b, h*64+e, t] ----
  float linv[8];
#pragma unroll
  for (int r = 0; r < 8; ++r) linv[r] = __builtin_amdgcn_rcpf(lrow[r]);
#pragma unroll
  for (int f = 0; f < 4; ++f) {
    float* orow =
        out + ((size_t)b * D_SZ + h * 64 + f * 16 + r16) * T_SZ + q0 + half * 8;
    v4f lo4, hi4;
#pragma unroll
    for (int i = 0; i < 4; ++i) {
      lo4[i] = o[f][i] * linv[i];
      hi4[i] = o[f][4 + i] * linv[4 + i];
    }
    *(v4f*)(orow)     = lo4;
    *(v4f*)(orow + 4) = hi4;
  }
}

// ---------------------------------------------------------------------------
extern "C" void kernel_launch(void* const* d_in, const int* in_sizes, int n_in,
                              void* d_out, int out_size, void* d_ws, size_t ws_size,
                              hipStream_t stream) {
  (void)in_sizes; (void)n_in; (void)out_size; (void)ws_size;
  const float* x   = (const float*)d_in[0]; // (4,1024,2048) f32
  const float* qkv = (const float*)d_in[1]; // (3,16,1024,64) f32
  float* out = (float*)d_out;               // (4,1024,2048) f32

  const size_t per = (size_t)B_SZ * H_SZ * T_SZ * DH; // 8,388,608 elems
  _Float16* qws = (_Float16*)d_ws;   // 16 MB
  _Float16* kws = qws + per;         // 16 MB
  _Float16* vt  = kws + per;         // 16 MB (transposed V)

  qkv_proj_norm_kernel<<<3 * B_SZ * H_SZ * (T_SZ / 128), 256, 0, stream>>>(
      x, qkv, qws, kws, vt);
  flash_attn_kernel<<<B_SZ * H_SZ * (T_SZ / 128), 256, 0, stream>>>(
      qws, kws, vt, out);
}
